// ScaledDotProductAttention_43327630082400
// MI455X (gfx1250) — compile-verified, matrix operand order checked by reading, not run on hardware
//
#include <hip/hip_runtime.h>

typedef __attribute__((ext_vector_type(16))) _Float16 v16h;
typedef __attribute__((ext_vector_type(8)))  _Float16 v8h;
typedef __attribute__((ext_vector_type(8)))  float    v8f;

#define BATCH 16
#define SEQ   2048
#define DHEAD 64
// log2(e) / temperature(=8): folds softmax base change + 1/sqrt(d_k) into Q
#define QSCALE 0.1803368801111204f

// Raw v_exp_f32 / v_rcp_f32: no libm subnormal/edge guards. Safe here because
// exp2 args are <= 0 (underflow->0 is the desired softmax behavior) and the
// softmax denominator is >= 1.
__device__ __forceinline__ float fexp2(float x) { return __builtin_amdgcn_exp2f(x); }
__device__ __forceinline__ float frcp(float x)  { return __builtin_amdgcn_rcpf(x); }

__device__ __forceinline__ v16h cat8(v8h lo, v8h hi) {
  return __builtin_shufflevector(lo, hi, 0,1,2,3,4,5,6,7,8,9,10,11,12,13,14,15);
}

// ---------------------------------------------------------------------------
// Pre-pass: f32 -> f16 conversion into workspace.
//   qh[b][s][d] = q * QSCALE   (A-matrix operand, scale folded in)
//   kh[b][s][d] = k            (B-matrix of QK^T: rows contiguous in d)
//   vt[b][d][s] = v[b][s][d]   (B-matrix of PV: rows contiguous in s)
// ---------------------------------------------------------------------------
__global__ void attn_pre(const float* __restrict__ q, const float* __restrict__ k,
                         const float* __restrict__ v, _Float16* __restrict__ qh,
                         _Float16* __restrict__ kh, _Float16* __restrict__ vt) {
  int i = blockIdx.x * blockDim.x + threadIdx.x;
  const int N = BATCH * SEQ * DHEAD;
  if (i >= N) return;
  int d = i & (DHEAD - 1);
  int s = (i >> 6) & (SEQ - 1);
  int b = i >> 17;                       // S*D = 2^17
  qh[i] = (_Float16)(q[i] * QSCALE);
  kh[i] = (_Float16)k[i];
  vt[(((size_t)b * DHEAD + d) << 11) + s] = (_Float16)v[i];
}

// One 16x16 score tile: scores[m][n] = sum_d Qh[m][d] * Kh[n][d]  (D=64 -> 2 WMMAs)
__device__ __forceinline__ v8f qk_tile(v16h aq0, v16h aq1, const _Float16* kp, int g) {
  v16h bk0 = cat8(*(const v8h*)(kp + g * 8),      *(const v8h*)(kp + 16 + g * 8));
  v16h bk1 = cat8(*(const v8h*)(kp + 32 + g * 8), *(const v8h*)(kp + 48 + g * 8));
  v8f acc = {0.f, 0.f, 0.f, 0.f, 0.f, 0.f, 0.f, 0.f};
  acc = __builtin_amdgcn_wmma_f32_16x16x32_f16(false, aq0, false, bk0, (short)0, acc, false, false);
  acc = __builtin_amdgcn_wmma_f32_16x16x32_f16(false, aq1, false, bk1, (short)0, acc, false, false);
  return acc;
}

// ---------------------------------------------------------------------------
// Fused attention: each wave32 owns one 16-row Q tile (2048 tiles total).
// Pass 1: LANE-LOCAL online softmax stats (no cross-lane ops in the loop),
//         one butterfly (m,l)-merge per wave at the end. Unroll 4 so the
//         scheduler can fill WMMA->VALU hazard slots with neighboring tiles.
// Pass 2: recompute scores (K/V are L2-resident f16, 8 MB total), normalize,
//         stream attn with non-temporal stores, accumulate P@V via WMMA.
// ---------------------------------------------------------------------------
__global__ void __launch_bounds__(128) attn_main(
    const _Float16* __restrict__ qh, const _Float16* __restrict__ kh,
    const _Float16* __restrict__ vt, float* __restrict__ out,
    float* __restrict__ attn) {
  __shared__ _Float16 ldsP[4][16 * 32];      // per-wave private P tile (1 KB each)

  const int lane = threadIdx.x & 31;
  const int wv   = threadIdx.x >> 5;
  const int tile = blockIdx.x * 4 + wv;      // 0..2047
  const int b    = tile >> 7;                // / (SEQ/16)
  const int qbase = (tile & 127) << 4;
  const int g    = lane >> 4;                // half-wave (A/B layout K-group)
  const int n15  = lane & 15;

  // --- Q A-tiles (16x32 f16 each); lane holds row m=n15, d runs per layout ---
  const _Float16* qp = qh + ((size_t)b * SEQ + qbase + n15) * DHEAD;
  v16h aq0 = cat8(*(const v8h*)(qp + g * 8),      *(const v8h*)(qp + 16 + g * 8));
  v16h aq1 = cat8(*(const v8h*)(qp + 32 + g * 8), *(const v8h*)(qp + 48 + g * 8));

  const _Float16* kpb = kh + (size_t)b * SEQ * DHEAD;

  // --- Pass 1: lane-local online max/sum (base-2 domain), zero DS traffic ----
  float m_ln[8], l_ln[8];
#pragma unroll
  for (int r = 0; r < 8; ++r) { m_ln[r] = -1e30f; l_ln[r] = 0.f; }

#pragma unroll 4
  for (int kt = 0; kt < SEQ / 16; ++kt) {
    const _Float16* kp = kpb + (size_t)((kt << 4) + n15) * DHEAD;
    v8f acc = qk_tile(aq0, aq1, kp, g);
#pragma unroll
    for (int r = 0; r < 8; ++r) {          // lane sees row r+8g, col n15+16*kt
      float t  = acc[r];
      float nm = fmaxf(m_ln[r], t);
      l_ln[r]  = l_ln[r] * fexp2(m_ln[r] - nm) + fexp2(t - nm);
      m_ln[r]  = nm;
    }
  }

  // Merge 16 lane-local (m,l) pairs per row: 4 butterfly steps, once per wave.
  float m_run[8], rl[8];
#pragma unroll
  for (int r = 0; r < 8; ++r) {
    float m = m_ln[r], l = l_ln[r];
#pragma unroll
    for (int off = 1; off < 16; off <<= 1) {
      float mo = __shfl_xor(m, off, 32);
      float lo = __shfl_xor(l, off, 32);
      float nm = fmaxf(m, mo);
      l = l * fexp2(m - nm) + lo * fexp2(mo - nm);
      m = nm;
    }
    m_run[r] = m;
    rl[r]    = frcp(l);     // l >= 1 always (row max contributes exp2(0) = 1)
  }

  // --- Pass 2: normalize + stream attn + accumulate P@V ----------------------
  v8f cz = {0.f, 0.f, 0.f, 0.f, 0.f, 0.f, 0.f, 0.f};
  v8f cacc[4] = {cz, cz, cz, cz};
  _Float16* lp = &ldsP[wv][0];
  float* attnRow = attn + ((size_t)b * SEQ + qbase + 8 * g) * SEQ + n15;
  const _Float16* vpb = vt + (size_t)b * DHEAD * SEQ;

#pragma unroll 2
  for (int ch = 0; ch < SEQ / 32; ++ch) {
#pragma unroll
    for (int sub = 0; sub < 2; ++sub) {
      const int kt = ch * 2 + sub;
      const _Float16* kp = kpb + (size_t)((kt << 4) + n15) * DHEAD;
      v8f acc = qk_tile(aq0, aq1, kp, g);
#pragma unroll
      for (int r = 0; r < 8; ++r) {
        float p = fexp2(acc[r] - m_run[r]) * rl[r];
        // attn is write-once, never re-read: non-temporal keeps K/V in L2
        __builtin_nontemporal_store(p, &attnRow[(size_t)r * SEQ + (kt << 4)]);
        lp[(r + 8 * g) * 32 + sub * 16 + n15] = (_Float16)p;      // C->A relayout
      }
    }
    asm volatile("s_wait_dscnt 0" ::: "memory");   // P tile visible before reload

    // P A-tile (16x32): lane row m=n15, key runs g*8 and 16+g*8
    v16h ap = cat8(*(const v8h*)(lp + n15 * 32 + g * 8),
                   *(const v8h*)(lp + n15 * 32 + 16 + g * 8));

    // V B-tiles from transposed Vt: 4 N-groups of 16 d-columns
#pragma unroll
    for (int t = 0; t < 4; ++t) {
      const _Float16* vr = vpb + (size_t)(t * 16 + n15) * SEQ + (ch << 5);
      v16h bv = cat8(*(const v8h*)(vr + g * 8), *(const v8h*)(vr + 16 + g * 8));
      cacc[t] = __builtin_amdgcn_wmma_f32_16x16x32_f16(false, ap, false, bv,
                                                       (short)0, cacc[t], false, false);
    }
  }

  // --- Output: D tile (m = r + 8g, n = 16t + n15) -----------------------------
  float* op = out + ((size_t)b * SEQ + qbase + 8 * g) * DHEAD + n15;
#pragma unroll
  for (int t = 0; t < 4; ++t)
#pragma unroll
    for (int r = 0; r < 8; ++r)
      __builtin_nontemporal_store(cacc[t][r], &op[(size_t)r * DHEAD + t * 16]);
}

// ---------------------------------------------------------------------------
extern "C" void kernel_launch(void* const* d_in, const int* in_sizes, int n_in,
                              void* d_out, int out_size, void* d_ws, size_t ws_size,
                              hipStream_t stream) {
  const float* q = (const float*)d_in[0];
  const float* k = (const float*)d_in[1];
  const float* v = (const float*)d_in[2];

  const size_t N = (size_t)BATCH * SEQ * DHEAD;        // 2,097,152
  float* out  = (float*)d_out;                         // [B,S,D]
  float* attn = (float*)d_out + N;                     // [B,S,S]

  _Float16* qh = (_Float16*)d_ws;                      // 3 * 4 MB f16 workspace
  _Float16* kh = qh + N;
  _Float16* vt = kh + N;

  attn_pre<<<(int)((N + 255) / 256), 256, 0, stream>>>(q, k, v, qh, kh, vt);

  const int tiles = BATCH * (SEQ / 16);                // 2048 wave-tiles
  attn_main<<<tiles / 4, 128, 0, stream>>>(qh, kh, vt, out, attn);
}
